// RCNN_NO_REGRESSOR_2121713844585
// MI455X (gfx1250) — matrix operand order repaired and checked by reading, not compiled, not run on hardware
//
#include <hip/hip_runtime.h>
#include <hip/hip_bf16.h>

// ---------------------------------------------------------------------------
// Types for CDNA5 WMMA (wave32): v_wmma_f32_16x16x32_bf16
// ---------------------------------------------------------------------------
typedef __attribute__((ext_vector_type(16))) __bf16 v16bf;
typedef __attribute__((ext_vector_type(8)))  float  v8f;

union BF16x16 {
    uint4        q[2];
    unsigned int u[8];
    v16bf        v;
};

__device__ __forceinline__ unsigned short f2bf(float x) {
    unsigned int u = __builtin_bit_cast(unsigned int, x);
    unsigned int r = u + 0x7FFFu + ((u >> 16) & 1u);   // round-to-nearest-even
    return (unsigned short)(r >> 16);
}
__device__ __forceinline__ float bf2f(unsigned short b) {
    return __builtin_bit_cast(float, (unsigned int)b << 16);
}

// Problem constants
#define Bn    64
#define Ln    512
#define Dn    300
#define DPAD  320          // padded row stride == tap segment length
#define Fn    128
#define KIM   900
#define KPAD  960          // 3 * DPAD, 30 chunks of 32
#define NCHUNK (KPAD / 32)
#define Nroi  2048
#define NCLS  8

// ---------------------------------------------------------------------------
// Prep 1: conv weights f32 [F, 3*300] -> bf16 [F, 960], k = tap*320 + d,
// zeros for d in [300,320). Matches the padded-LDS im2col ordering.
// ---------------------------------------------------------------------------
__global__ __launch_bounds__(256) void prep_conv_w(const float* __restrict__ cw,
                                                   unsigned short* __restrict__ wpk) {
    int idx = blockIdx.x * 256 + threadIdx.x;
    if (idx >= Fn * KPAD) return;
    int f   = idx / KPAD;
    int k   = idx - f * KPAD;
    int tap = k / DPAD;
    int d   = k - tap * DPAD;
    float v = (d < Dn) ? cw[f * KIM + tap * Dn + d] : 0.0f;
    wpk[idx] = f2bf(v);
}

// ---------------------------------------------------------------------------
// Prep 2: collapse repeat(4) + fc1 + cls into Weff[8,128] (+pad to 16) & beff.
//   Weff[c,f] = sum_j cls_w[c,j] * sum_p fc1_w[j,4f+p]
//   beff[c]   = cls_b[c] + sum_j cls_w[c,j]*fc1_b[j]
// Weff stored B-fragment-native: elem((kk,h,n,j)) at ((kk*2+h)*16+n)*16 + j
//   -> value Weff[c=n][f = kk*32 + 16h + j]; one contiguous 32B run per lane.
// ---------------------------------------------------------------------------
__global__ __launch_bounds__(256) void prep_fc(const float* __restrict__ fc1_w,
                                               const float* __restrict__ fc1_b,
                                               const float* __restrict__ cls_w,
                                               const float* __restrict__ cls_b,
                                               unsigned short* __restrict__ weff2,
                                               float* __restrict__ beff) {
    const int FF = Fn * 4;   // 512
    int tid = threadIdx.x;
    for (int o = tid; o < Fn * 16; o += 256) {
        int j  = o & 15;
        int n  = (o >> 4) & 15;
        int h  = (o >> 8) & 1;
        int kk = o >> 9;
        int f  = kk * 32 + 16 * h + j;
        int c  = n;
        float s = 0.0f;
        if (c < NCLS) {
            const float* cw = cls_w + c * FF;
            for (int jj = 0; jj < FF; ++jj) {
                const float* w1 = fc1_w + (size_t)jj * FF + 4 * f;
                s += cw[jj] * (w1[0] + w1[1] + w1[2] + w1[3]);
            }
        }
        weff2[o] = f2bf(s);
    }
    if (tid < 16) {
        float s = 0.0f;
        if (tid < NCLS) {
            const float* cw = cls_w + tid * FF;
            s = cls_b[tid];
            for (int jj = 0; jj < FF; ++jj) s += cw[jj] * fc1_b[jj];
        }
        beff[tid] = s;
    }
}

// ---------------------------------------------------------------------------
// Conv as im2col GEMM via WMMA bf16, A = weights (global), B = activations (LDS).
// D: M = features, N = l positions -> each D lane holds 8 consecutive f at one l,
// so the epilogue is a single b128 store per tile (no LDS transpose).
// grid = (L/64, B), block = 256 (8 waves). Wave (fg,lg) owns a 32x32 output tile:
// 2 A frags + 2 B frags -> 4 WMMAs per K chunk (30 chunks, fully unrolled),
// with explicit register double-buffering: chunk kk+1's fragments are loaded
// before chunk kk's WMMAs issue, so waits cover a full chunk of XDL work.
// ---------------------------------------------------------------------------
__global__ __launch_bounds__(256, 1) void conv_gemm(const float* __restrict__ sent,
                                                    const unsigned short* __restrict__ wpk,
                                                    const float* __restrict__ conv_b,
                                                    unsigned short* __restrict__ feat) {
    __shared__ __align__(16) unsigned short ldsA[66 * DPAD];   // 42,240 bytes

    const int b      = blockIdx.y;
    const int lstart = blockIdx.x * 64;
    const int tid    = threadIdx.x;
    const int wave   = tid >> 5;
    const int lane   = tid & 31;

    // Stage rows lstart-1 .. lstart+64 as bf16, stride 320, zero halo+pad cols.
    for (int g = tid; g < 66 * 80; g += 256) {
        int r    = g / 80;
        int c4   = (g - r * 80) * 4;
        int row  = lstart - 1 + r;
        uint2 packed = make_uint2(0u, 0u);
        if (c4 < Dn && row >= 0 && row < Ln) {
            const float4 v = *(const float4*)(sent + ((size_t)b * Ln + row) * Dn + c4);
            packed.x = (unsigned)f2bf(v.x) | ((unsigned)f2bf(v.y) << 16);
            packed.y = (unsigned)f2bf(v.z) | ((unsigned)f2bf(v.w) << 16);
        }
        *(uint2*)(ldsA + r * DPAD + c4) = packed;
    }
    __syncthreads();

    const int n  = lane & 15;
    const int h  = lane >> 4;
    const int m0 = lane & 15;
    const int fg = (wave & 3) * 32;        // feature base of this wave
    const int lg = (wave >> 2) * 32;       // position base of this wave

    // A (weights): rows f, per-lane 16B runs at kk*32 + 8h and +16.
    const unsigned short* arow0 = wpk + (size_t)(fg + m0) * KPAD + 8 * h;
    const unsigned short* arow1 = arow0 + 16 * KPAD;
    // B (activations): LDS rows l, per-lane 16B runs at kk*32 + 8h and +16.
    const unsigned short* brow0 = ldsA + (lg + n) * DPAD + 8 * h;
    const unsigned short* brow1 = brow0 + 16 * DPAD;

    auto loadf = [&](int kk, BF16x16& A0, BF16x16& A1, BF16x16& B0, BF16x16& B1) {
        A0.q[0] = *(const uint4*)(arow0 + kk * 32);
        A0.q[1] = *(const uint4*)(arow0 + kk * 32 + 16);
        A1.q[0] = *(const uint4*)(arow1 + kk * 32);
        A1.q[1] = *(const uint4*)(arow1 + kk * 32 + 16);
        B0.q[0] = *(const uint4*)(brow0 + kk * 32);
        B0.q[1] = *(const uint4*)(brow0 + kk * 32 + 16);
        B1.q[0] = *(const uint4*)(brow1 + kk * 32);
        B1.q[1] = *(const uint4*)(brow1 + kk * 32 + 16);
    };

    v8f acc00 = {}, acc01 = {}, acc10 = {}, acc11 = {};
    BF16x16 A0[2], A1[2], B0[2], B1[2];
    loadf(0, A0[0], A1[0], B0[0], B1[0]);

#pragma unroll
    for (int kk = 0; kk < NCHUNK; ++kk) {
        const int cur = kk & 1, nxt = cur ^ 1;
        if (kk + 1 < NCHUNK) loadf(kk + 1, A0[nxt], A1[nxt], B0[nxt], B1[nxt]);
        acc00 = __builtin_amdgcn_wmma_f32_16x16x32_bf16(false, A0[cur].v, false, B0[cur].v, (short)0, acc00, false, false);
        acc01 = __builtin_amdgcn_wmma_f32_16x16x32_bf16(false, A0[cur].v, false, B1[cur].v, (short)0, acc01, false, false);
        acc10 = __builtin_amdgcn_wmma_f32_16x16x32_bf16(false, A1[cur].v, false, B0[cur].v, (short)0, acc10, false, false);
        acc11 = __builtin_amdgcn_wmma_f32_16x16x32_bf16(false, A1[cur].v, false, B1[cur].v, (short)0, acc11, false, false);
    }

    // Epilogue: +bias, ReLU, pack 8 bf16 -> one b128 store per tile.
    v8f accs[2][2] = {{acc00, acc01}, {acc10, acc11}};
#pragma unroll
    for (int mt = 0; mt < 2; ++mt) {
        const int fbase = fg + mt * 16 + 8 * h;      // 8 consecutive features
        const float4 bA = *(const float4*)(conv_b + fbase);
        const float4 bB = *(const float4*)(conv_b + fbase + 4);
        float bias[8] = {bA.x, bA.y, bA.z, bA.w, bB.x, bB.y, bB.z, bB.w};
#pragma unroll
        for (int nt = 0; nt < 2; ++nt) {
            v8f a = accs[mt][nt];
            unsigned int w[4];
#pragma unroll
            for (int p = 0; p < 4; ++p) {
                float v0 = a[2 * p]     + bias[2 * p];
                float v1 = a[2 * p + 1] + bias[2 * p + 1];
                v0 = v0 > 0.0f ? v0 : 0.0f;
                v1 = v1 > 0.0f ? v1 : 0.0f;
                w[p] = (unsigned)f2bf(v0) | ((unsigned)f2bf(v1) << 16);
            }
            const int l = lstart + lg + nt * 16 + n;
            uint4* dst = (uint4*)(feat + ((size_t)b * Ln + l) * Fn + fbase);
            *dst = make_uint4(w[0], w[1], w[2], w[3]);
        }
    }
}

// ---------------------------------------------------------------------------
// ROI max-pool over span [x1,x2): block = 64 threads, 2 features per thread.
// ---------------------------------------------------------------------------
__global__ __launch_bounds__(64) void roi_pool(const unsigned short* __restrict__ feat,
                                               const int* __restrict__ rois,
                                               const int* __restrict__ ridx,
                                               unsigned short* __restrict__ pooled) {
    int nidx = blockIdx.x;
    int t    = threadIdx.x;
    int b    = ridx[nidx];
    int x1   = rois[2 * nidx + 0];
    int x2   = rois[2 * nidx + 1];
    float m0 = -3.4e38f, m1 = -3.4e38f;
    const unsigned int* base = (const unsigned int*)(feat + ((size_t)b * Ln) * Fn) + t;
    for (int l = x1; l < x2; ++l) {
        unsigned int p = base[(size_t)l * (Fn / 2)];
        m0 = fmaxf(m0, bf2f((unsigned short)(p & 0xFFFFu)));
        m1 = fmaxf(m1, bf2f((unsigned short)(p >> 16)));
    }
    unsigned int outp = (unsigned)f2bf(m0) | ((unsigned)f2bf(m1) << 16);
    ((unsigned int*)(pooled + (size_t)nidx * Fn))[t] = outp;
}

// ---------------------------------------------------------------------------
// Final collapsed GEMM: [2048 x 128] @ [128 x 16(pad of 8)] via WMMA bf16.
// ---------------------------------------------------------------------------
__global__ __launch_bounds__(256) void final_gemm(const unsigned short* __restrict__ pooled,
                                                  const unsigned short* __restrict__ weff2,
                                                  const float* __restrict__ beff,
                                                  float* __restrict__ out) {
    const int tid   = threadIdx.x;
    const int wave  = tid >> 5;
    const int lane  = tid & 31;
    const int mbase = (blockIdx.x * 8 + wave) * 16;
    const int n  = lane & 15;
    const int h  = lane >> 4;
    const int m0 = lane & 15;

    v8f acc = {};
    const unsigned short* arow = pooled + (size_t)(mbase + m0) * Fn + 16 * h;

#pragma unroll
    for (int kk = 0; kk < Fn / 32; ++kk) {
        BF16x16 af, bf;
        af.q[0] = *(const uint4*)(arow + kk * 32);
        af.q[1] = *(const uint4*)(arow + kk * 32 + 8);
        const unsigned short* bb = weff2 + (size_t)(((kk * 2 + h) * 16 + n) * 16);
        bf.q[0] = *(const uint4*)(bb);
        bf.q[1] = *(const uint4*)(bb + 8);
        acc = __builtin_amdgcn_wmma_f32_16x16x32_bf16(
            false, af.v, false, bf.v, (short)0, acc, false, false);
    }

    if (n < NCLS) {
        float bb = beff[n];
#pragma unroll
        for (int i = 0; i < 8; ++i) {
            int m = mbase + i + 8 * h;
            out[(size_t)m * NCLS + n] = acc[i] + bb;
        }
    }
}

// ---------------------------------------------------------------------------
// Host-side launcher
// ---------------------------------------------------------------------------
extern "C" void kernel_launch(void* const* d_in, const int* in_sizes, int n_in,
                              void* d_out, int out_size, void* d_ws, size_t ws_size,
                              hipStream_t stream) {
    (void)in_sizes; (void)n_in; (void)out_size; (void)ws_size;
    const float* sent   = (const float*)d_in[0];
    const int*   rois   = (const int*)d_in[1];
    const int*   ridx   = (const int*)d_in[2];
    const float* conv_w = (const float*)d_in[3];
    const float* conv_b = (const float*)d_in[4];
    const float* fc1_w  = (const float*)d_in[5];
    const float* fc1_b  = (const float*)d_in[6];
    const float* cls_w  = (const float*)d_in[7];
    const float* cls_b  = (const float*)d_in[8];
    float* out = (float*)d_out;
    char*  ws  = (char*)d_ws;

    // Workspace layout (256-aligned)
    unsigned short* wpk    = (unsigned short*)(ws + 0);          // 128*960 bf16 = 245760 B
    unsigned short* weff2  = (unsigned short*)(ws + 245760);     // 2048 bf16   = 4096 B
    float*          beff   = (float*)(ws + 249856);              // 16 f32
    unsigned short* feat   = (unsigned short*)(ws + 250112);     // 64*512*128 bf16 = 8388608 B
    unsigned short* pooled = (unsigned short*)(ws + 8638720);    // 2048*128 bf16 = 524288 B

    prep_conv_w<<<(Fn * KPAD + 255) / 256, 256, 0, stream>>>(conv_w, wpk);
    prep_fc<<<1, 256, 0, stream>>>(fc1_w, fc1_b, cls_w, cls_b, weff2, beff);
    conv_gemm<<<dim3(Ln / 64, Bn), 256, 0, stream>>>(sent, wpk, conv_b, feat);
    roi_pool<<<Nroi, 64, 0, stream>>>(feat, rois, ridx, pooled);
    final_gemm<<<Nroi / (8 * 16), 256, 0, stream>>>(pooled, weff2, beff, out);
}
